// WeightedBoxFusion_3264175145227
// MI455X (gfx1250) — compile-verified, matrix operand order checked by reading, not run on hardware
//
#include <hip/hip_runtime.h>
#include <hip/hip_bf16.h>
#include <stdint.h>
#include <float.h>

// ---------------------------------------------------------------------------
// Weighted Box Fusion on MI455X (gfx1250).
//
// Dataset ~1.5MB => entirely L2/LDS resident; HBM irrelevant. The cost is the
// 12288-step *serial* clustering scan. Design:
//  * single-workgroup scan, cluster reps+labels in 245KB of LDS (CDNA5 WGP
//    has 320KB LDS; this layout is impossible on CDNA4),
//  * wave32 __shfl_xor argmax reduction + 32-wave LDS reduce per step,
//  * sorted step-stream staged to LDS in 512-step chunks via
//    global_load_async_to_lds_b128 / s_wait_asynccnt (gfx1250 async path),
//  * weighted scores fs = diag(w) x S computed with v_wmma_f32_16x16x32_f16,
//  * both argsorts as single-launch 16K-element in-LDS bitonic sorts with an
//    index tiebreak (total order == stable argsort semantics).
// ---------------------------------------------------------------------------

#define KMOD 3
#define NBOX 4096
#define TBOX (KMOD * NBOX)  // 12288
#define POW2 16384          // bitonic size (power of two >= TBOX)
#define IOU_THR 0.55f
#define EPSV 1e-8f

typedef __attribute__((ext_vector_type(16))) _Float16 v16h;
typedef __attribute__((ext_vector_type(8)))  float    v8f;

// ---------------- gfx1250 async global->LDS helpers (inline asm; portable
// across ROCm7.2/clang-22 and clang-23 toolchains, bypasses builtin arity
// differences). LDS byte address = low 32 bits of the generic pointer.
__device__ __forceinline__ void async_copy_b128(void* lds_ptr, const void* gptr) {
    uint32_t lds_addr = (uint32_t)(uintptr_t)lds_ptr;
    uint64_t gaddr    = (uint64_t)(uintptr_t)gptr;
    asm volatile("global_load_async_to_lds_b128 %0, %1, off"
                 :: "v"(lds_addr), "v"(gaddr) : "memory");
}
__device__ __forceinline__ void wait_async0() {
    asm volatile("s_wait_asynccnt 0x0" ::: "memory");
}

// ---------------- Phase 1: fs = diag(w) * scores, one WMMA per 16-col tile.
// A (16x32 f16): A[m][k] = w[m] iff k==m (m<3).  Lane m<16 holds row m; f16
// A-layout maps K=e to element e (e<8), so a[lane]=w[lane] for lane<3.
// B (32x16 f16): lane j<16 holds column j, element k = scores[k][col].
// D (16x16 f32): lane j<16, element r = D[r][j] = w[r]*scores[r][col].
__global__ void fs_wmma_kernel(const float* __restrict__ scores,
                               const float* __restrict__ weights,
                               float* __restrict__ fs) {
    int gtid = blockIdx.x * blockDim.x + threadIdx.x;
    int wave = gtid >> 5;
    int lane = gtid & 31;
    int tile = wave;              // 256 waves == 256 tiles of 16 columns
    v16h a = {};
    v16h b = {};
    if (lane < KMOD) a[lane] = (_Float16)weights[lane];
    if (lane < 16) {
        int col = tile * 16 + lane;
        b[0] = (_Float16)scores[0 * NBOX + col];
        b[1] = (_Float16)scores[1 * NBOX + col];
        b[2] = (_Float16)scores[2 * NBOX + col];
    }
    v8f c = {};
    c = __builtin_amdgcn_wmma_f32_16x16x32_f16(false, a, false, b,
                                               (short)0, c, false, false);
    if (lane < 16) {
        int col = tile * 16 + lane;
        fs[0 * NBOX + col] = c[0];
        fs[1 * NBOX + col] = c[1];
        fs[2 * NBOX + col] = c[2];
    }
}

// ---------------- scratch zeroing (harness poisons ws; re-zero every call).
__global__ void zero_kernel(float* p, int n) {
    int i = blockIdx.x * blockDim.x + threadIdx.x;
    if (i < n) p[i] = 0.0f;
}

// ---------------- sort-key init: key = fs (pad -FLT_MAX), val = index.
__global__ void fill_keys_kernel(const float* __restrict__ fs,
                                 float* __restrict__ key, int* __restrict__ val) {
    int i = blockIdx.x * blockDim.x + threadIdx.x;
    if (i >= POW2) return;
    key[i] = (i < TBOX) ? fs[i] : -FLT_MAX;
    val[i] = i;
}

// ---------------- in-LDS bitonic sort of 16384 (key,val) pairs, descending
// key with ascending-index tiebreak (total order -> deterministic, matches
// stable argsort(-key)).  128KB dynamic LDS, one workgroup of 1024 threads.
__device__ __forceinline__ bool precedes(float ka, int va, float kb, int vb) {
    return (ka > kb) || (ka == kb && va < vb);
}

__global__ __launch_bounds__(1024) void bitonic_kernel(float* key, int* val) {
    extern __shared__ char smem[];
    float* k = (float*)smem;
    int*   v = (int*)(smem + (size_t)POW2 * 4);
    int tid = threadIdx.x;
    for (int i = tid; i < POW2; i += 1024) { k[i] = key[i]; v[i] = val[i]; }
    __syncthreads();
    for (int size = 2; size <= POW2; size <<= 1) {
        for (int stride = size >> 1; stride > 0; stride >>= 1) {
            for (int p = tid; p < POW2 / 2; p += 1024) {
                int i = (p / stride) * (stride << 1) + (p % stride);
                int j = i + stride;
                bool finalDir = ((i & size) == 0);
                bool ji = precedes(k[j], v[j], k[i], v[i]);  // j should come first
                if (finalDir ? ji : !ji) {
                    float tk = k[i]; k[i] = k[j]; k[j] = tk;
                    int   tv = v[i]; v[i] = v[j]; v[j] = tv;
                }
            }
            __syncthreads();
        }
    }
    for (int i = tid; i < POW2; i += 1024) { key[i] = k[i]; val[i] = v[i]; }
}

// ---------------- gather sorted tuples into two float4 streams:
// sbox[r] = box, sdat[r] = (weighted score, label, model, 0).
__global__ void gather_kernel(const float* __restrict__ key, const int* __restrict__ val,
                              const float* __restrict__ boxes, const int* __restrict__ labels,
                              float4* __restrict__ sbox, float4* __restrict__ sdat) {
    int r = blockIdx.x * blockDim.x + threadIdx.x;
    if (r >= TBOX) return;
    int idx = val[r];
    int m = idx / NBOX;
    float4 b;
    b.x = boxes[idx * 4 + 0]; b.y = boxes[idx * 4 + 1];
    b.z = boxes[idx * 4 + 2]; b.w = boxes[idx * 4 + 3];
    sbox[r] = b;
    float4 d;
    d.x = key[r]; d.y = (float)labels[idx]; d.z = (float)m; d.w = 0.0f;
    sdat[r] = d;
}

// ---------------- the serial WBF clustering scan.  One workgroup, 1024 thr.
// Dynamic LDS: rep[T] float4 (192KB) | lab[T] int (48KB) | stage[1024] float4
// (16KB: 512 boxes + 512 dats per chunk).  Accumulators (touched at a single
// slot per step) live in global scratch (L2-resident).
#define CHUNK 512
__global__ __launch_bounds__(1024) void scan_kernel(
        const float4* __restrict__ sbox, const float4* __restrict__ sdat,
        float* __restrict__ wbox, float* __restrict__ wsum,
        float* __restrict__ ssum, float* __restrict__ cntArr,
        int* __restrict__ mbits, int* __restrict__ glab, int* __restrict__ countOut) {
    extern __shared__ char smem[];
    float4* rep   = (float4*)smem;                                     // TBOX
    int*    lab   = (int*)(smem + (size_t)TBOX * 16);                  // TBOX
    float4* stage = (float4*)(smem + (size_t)TBOX * 16 + (size_t)TBOX * 4); // 1024

    __shared__ float redIou[32];
    __shared__ int   redIdx[32];
    __shared__ int   sCount;
    __shared__ int   sIdx;

    int tid  = threadIdx.x;
    int lane = tid & 31;
    int wid  = tid >> 5;
    if (tid == 0) sCount = 0;

    for (int base = 0; base < TBOX; base += CHUNK) {
        // async-stage this chunk's step data (512 box float4 + 512 dat float4)
        const void* g = (tid < CHUNK) ? (const void*)&sbox[base + tid]
                                      : (const void*)&sdat[base + tid - CHUNK];
        async_copy_b128(&stage[tid], g);
        wait_async0();
        __syncthreads();

        for (int s = 0; s < CHUNK; ++s) {
            float4 box = stage[s];
            float4 dat = stage[CHUNK + s];
            float sc = dat.x;
            int   lb = (int)dat.y;
            int   mi = (int)dat.z;
            int count = sCount;

            float bi = -1.0f;
            int   br = count;
            float a1 = (box.z - box.x) * (box.w - box.y);
            for (int r = tid; r < count; r += 1024) {
                if (lab[r] == lb) {
                    float4 rp = rep[r];
                    float x1 = fmaxf(box.x, rp.x), y1 = fmaxf(box.y, rp.y);
                    float x2 = fminf(box.z, rp.z), y2 = fminf(box.w, rp.w);
                    float inter = fmaxf(x2 - x1, 0.f) * fmaxf(y2 - y1, 0.f);
                    float a2 = (rp.z - rp.x) * (rp.w - rp.y);
                    float iou = inter / (a1 + a2 - inter + EPSV);
                    if (iou > bi || (iou == bi && r < br)) { bi = iou; br = r; }
                }
            }
            // wave32 argmax reduce (lowest index wins ties, like jnp.argmax)
            for (int off = 16; off; off >>= 1) {
                float oi = __shfl_xor(bi, off, 32);
                int   orr = __shfl_xor(br, off, 32);
                if (oi > bi || (oi == bi && orr < br)) { bi = oi; br = orr; }
            }
            if (lane == 0) { redIou[wid] = bi; redIdx[wid] = br; }
            __syncthreads();

            if (tid == 0) {
                float best = -1.0f; int bidx = count;
                for (int w = 0; w < 32; ++w) {
                    float x = redIou[w]; int y = redIdx[w];
                    if (x > best || (x == best && y < bidx)) { best = x; bidx = y; }
                }
                bool match = best > IOU_THR;
                int idx = match ? bidx : count;
                if (!match) { rep[count] = box; lab[count] = lb; sCount = count + 1; }
                sIdx = idx;
            }
            __syncthreads();

            // accumulator update: fixed lane per component => in-order per lane
            int idx = sIdx;
            if (tid < 4)       wbox[idx * 4 + tid] += ((const float*)&box)[tid] * sc;
            else if (tid == 4) wsum[idx]  += sc;
            else if (tid == 5) ssum[idx]  += sc;
            else if (tid == 6) cntArr[idx] += 1.0f;
            else if (tid == 7) mbits[idx] |= (1 << mi);
        }
        __syncthreads();
    }
    if (tid == 0) *countOut = sCount;
    __syncthreads();
    for (int i = tid; i < TBOX; i += 1024) glab[i] = (i < sCount) ? lab[i] : 0;
}

// ---------------- fuse clusters, build second-sort keys.
__global__ void finalize_kernel(const float* __restrict__ wbox, const float* __restrict__ wsum,
                                const float* __restrict__ ssum, const float* __restrict__ cntArr,
                                const int* __restrict__ mbits, const int* __restrict__ glab,
                                const int* __restrict__ countOut,
                                float4* __restrict__ fbox, int* __restrict__ flab,
                                float* __restrict__ key2, int* __restrict__ val2) {
    int i = blockIdx.x * blockDim.x + threadIdx.x;
    if (i >= POW2) return;
    if (i < TBOX) {
        int count = *countOut;
        bool live = (i < count);
        float ws_ = wsum[i] + EPSV;
        float4 fb;
        fb.x = wbox[i * 4 + 0] / ws_; fb.y = wbox[i * 4 + 1] / ws_;
        fb.z = wbox[i * 4 + 2] / ws_; fb.w = wbox[i * 4 + 3] / ws_;
        float avg  = ssum[i] / fmaxf(cntArr[i], 1.0f);
        float nmod = (float)__popc(mbits[i]);
        float fsc  = avg * (fminf(nmod, (float)KMOD) / (float)KMOD);
        if (!live) { fsc = -1.0f; fb.x = fb.y = fb.z = fb.w = 0.0f; }
        fbox[i] = fb;
        flab[i] = live ? glab[i] : 0;
        key2[i] = fsc;
        val2[i] = i;
    } else {
        key2[i] = -FLT_MAX;
        val2[i] = i;
    }
}

// ---------------- emit outputs in final sorted order.
// d_out layout: boxes[T*4] | scores[T] | labels[T] (labels written as floats).
__global__ void scatter_kernel(const float* __restrict__ key2, const int* __restrict__ val2,
                               const float4* __restrict__ fbox, const int* __restrict__ flab,
                               float* __restrict__ out) {
    int r = blockIdx.x * blockDim.x + threadIdx.x;
    if (r >= TBOX) return;
    int j = val2[r];
    float4 b = fbox[j];
    out[r * 4 + 0] = b.x; out[r * 4 + 1] = b.y;
    out[r * 4 + 2] = b.z; out[r * 4 + 3] = b.w;
    out[TBOX * 4 + r] = key2[r];
    out[TBOX * 5 + r] = (float)flab[j];
}

// ---------------- workspace layout (bytes), ~1.39MB total.
#define O_FS    ((size_t)0)
#define O_KEY1  (O_FS   + (size_t)TBOX * 4)
#define O_VAL1  (O_KEY1 + (size_t)POW2 * 4)
#define O_SBOX  (O_VAL1 + (size_t)POW2 * 4)
#define O_SDAT  (O_SBOX + (size_t)TBOX * 16)
#define O_WBOX  (O_SDAT + (size_t)TBOX * 16)
#define O_WSUM  (O_WBOX + (size_t)TBOX * 16)
#define O_SSUM  (O_WSUM + (size_t)TBOX * 4)
#define O_CNT   (O_SSUM + (size_t)TBOX * 4)
#define O_MB    (O_CNT  + (size_t)TBOX * 4)
#define O_CN    (O_MB   + (size_t)TBOX * 4)
#define O_GLAB  (O_CN   + 16)
#define O_FBOX  (O_GLAB + (size_t)TBOX * 4)
#define O_FSC_UNUSED 0
#define O_FLAB  (O_FBOX + (size_t)TBOX * 16)
#define O_KEY2  (O_FLAB + (size_t)TBOX * 4)
#define O_VAL2  (O_KEY2 + (size_t)POW2 * 4)

extern "C" void kernel_launch(void* const* d_in, const int* in_sizes, int n_in,
                              void* d_out, int out_size, void* d_ws, size_t ws_size,
                              hipStream_t stream) {
    (void)in_sizes; (void)n_in; (void)out_size; (void)ws_size;
    const float* boxes   = (const float*)d_in[0];
    const float* scores  = (const float*)d_in[1];
    const int*   labels  = (const int*)d_in[2];
    const float* weights = (const float*)d_in[3];
    float* out = (float*)d_out;

    char* ws = (char*)d_ws;
    float*  fs    = (float*)(ws + O_FS);
    float*  key1  = (float*)(ws + O_KEY1);
    int*    val1  = (int*)  (ws + O_VAL1);
    float4* sbox  = (float4*)(ws + O_SBOX);
    float4* sdat  = (float4*)(ws + O_SDAT);
    float*  wbox  = (float*)(ws + O_WBOX);
    float*  wsum  = (float*)(ws + O_WSUM);
    float*  ssum  = (float*)(ws + O_SSUM);
    float*  cnt   = (float*)(ws + O_CNT);
    int*    mbits = (int*)  (ws + O_MB);
    int*    cn    = (int*)  (ws + O_CN);
    int*    glab  = (int*)  (ws + O_GLAB);
    float4* fbox  = (float4*)(ws + O_FBOX);
    int*    flab  = (int*)  (ws + O_FLAB);
    float*  key2  = (float*)(ws + O_KEY2);
    int*    val2  = (int*)  (ws + O_VAL2);

    // 1. re-zero accumulators (wbox..countOut contiguous): T*8+4 floats
    int nz = (int)((O_GLAB - O_WBOX) / 4);
    zero_kernel<<<(nz + 255) / 256, 256, 0, stream>>>(wbox, nz);

    // 2. weighted scores via WMMA (256 waves, one 16-col tile each)
    fs_wmma_kernel<<<64, 128, 0, stream>>>(scores, weights, fs);

    // 3. first argsort (descending fs, stable)
    fill_keys_kernel<<<POW2 / 256, 256, 0, stream>>>(fs, key1, val1);
    bitonic_kernel<<<1, 1024, (size_t)POW2 * 8, stream>>>(key1, val1);

    // 4. gather sorted tuples
    gather_kernel<<<TBOX / 256, 256, 0, stream>>>(key1, val1, boxes, labels, sbox, sdat);

    // 5. serial clustering scan (reps in 245KB LDS + 16KB async stage)
    size_t scanLds = (size_t)TBOX * 16 + (size_t)TBOX * 4 + (size_t)1024 * 16; // 262144
    scan_kernel<<<1, 1024, scanLds, stream>>>(sbox, sdat, wbox, wsum, ssum, cnt,
                                              mbits, glab, cn);

    // 6. fuse + second argsort keys
    finalize_kernel<<<POW2 / 256, 256, 0, stream>>>(wbox, wsum, ssum, cnt, mbits,
                                                    glab, cn, fbox, flab, key2, val2);
    bitonic_kernel<<<1, 1024, (size_t)POW2 * 8, stream>>>(key2, val2);

    // 7. emit
    scatter_kernel<<<TBOX / 256, 256, 0, stream>>>(key2, val2, fbox, flab, out);
}
